// EncoderBlock_64407329571215
// MI455X (gfx1250) — matrix operand order repaired and checked by reading, not compiled, not run on hardware
//
#include <hip/hip_runtime.h>
#include <math.h>

// ---------------- types ----------------
typedef __attribute__((ext_vector_type(16))) __bf16 v16bf;
typedef __attribute__((ext_vector_type(8)))  float  v8f;
typedef __attribute__((ext_vector_type(4)))  unsigned int v4u;
typedef __attribute__((ext_vector_type(8)))  int v8i;
typedef __attribute__((ext_vector_type(4)))  int v4i;

#define EMBED 768
#define HEADS 12
#define HDIM  64
#define DFF   3072
#define SEQ   2048
#define BATCH 4
#define ROWS  (BATCH*SEQ)   // 8192

// ---------------- bf16 helpers (bit-exact RNE) ----------------
__device__ __forceinline__ __bf16 f2bf(float f) {
  unsigned u = __builtin_bit_cast(unsigned, f);
  unsigned r = u + 0x7FFFu + ((u >> 16) & 1u);
  unsigned short h = (unsigned short)(r >> 16);
  return __builtin_bit_cast(__bf16, h);
}
__device__ __forceinline__ __bf16 u16bf(unsigned short h) {
  return __builtin_bit_cast(__bf16, h);
}

// ---------------- WMMA wrapper ----------------
__device__ __forceinline__ v8f wmma_bf16(v16bf a, v16bf b, v8f c) {
  return __builtin_amdgcn_wmma_f32_16x16x32_bf16(
      /*neg_a=*/false, a, /*neg_b=*/false, b,
      /*c_mod=*/(short)0, c, /*reuse_a=*/false, /*reuse_b=*/false);
}

// ---- fragment loaders (layouts per CDNA5 ISA §7.12.2, wave32) ----
// A: 16x32 (MxK) bf16, row-major in LDS with leading dim ld.
__device__ __forceinline__ v16bf load_a_frag(const __bf16* base, int ld) {
  int lane = threadIdx.x & 31;
  int r  = lane & 15;
  int ko = (lane >> 4) * 8;
  const __bf16* p = base + r * ld + ko;
  v16bf a;
#pragma unroll
  for (int i = 0; i < 8; ++i) { a[i] = p[i]; a[i + 8] = p[16 + i]; }
  return a;
}
// B: 32x16 (KxN) bf16, stored TRANSPOSED in LDS as Bt[n][k] with leading dim ld.
__device__ __forceinline__ v16bf load_b_frag(const __bf16* baseT, int ld) {
  int lane = threadIdx.x & 31;
  int n  = lane & 15;
  int ko = (lane >> 4) * 8;
  const __bf16* p = baseT + n * ld + ko;
  v16bf b;
#pragma unroll
  for (int i = 0; i < 8; ++i) { b[i] = p[i]; b[i + 8] = p[16 + i]; }
  return b;
}

// ---------------- Tensor Data Mover: 2-D bf16 tile -> LDS ----------------
// Descriptor packing per CDNA5 ISA.
// 6-arg builtin variant (clang-23 / therock HIP headers).
__device__ __forceinline__ void tdm_load_2d_bf16(unsigned ldsOff, const void* gaddr,
                                                 unsigned tensor_d0, unsigned tensor_d1,
                                                 unsigned stride0,   // elems between dim-1 lines
                                                 unsigned tile_d0, unsigned tile_d1,
                                                 unsigned w0flags) {
  unsigned long long ga = (unsigned long long)(uintptr_t)gaddr;
  v4u g0;
  g0[0] = 1u;                                                    // count=1 (valid), user mode
  g0[1] = ldsOff;                                                // lds_addr (bytes)
  g0[2] = (unsigned)ga;                                          // global_addr[31:0]
  g0[3] = (unsigned)((ga >> 32) & 0x01FFFFFFu) | (2u << 30);     // global_addr[56:32] | type=2
  v8i g1;
  g1[0] = (int)w0flags;                                          // mask=0, data_size, pad cfg
  g1[1] = (int)((tensor_d0 & 0xFFFFu) << 16);                    // atomic_barrier=0 | dim0 lo
  g1[2] = (int)((tensor_d0 >> 16) | ((tensor_d1 & 0xFFFFu) << 16));
  g1[3] = (int)((tensor_d1 >> 16) | (tile_d0 << 16));            // dim1 hi | tile_dim0
  g1[4] = (int)tile_d1;                                          // tile_dim1 | tile_dim2=0
  g1[5] = (int)stride0;                                          // tensor_dim0_stride[31:0]
  g1[6] = 0;                                                     // stride0 hi | stride1 lo
  g1[7] = 0;                                                     // stride1 hi
  v4i gz4  = {0, 0, 0, 0};
  v8i gz8  = {0, 0, 0, 0, 0, 0, 0, 0};
  __builtin_amdgcn_tensor_load_to_lds(g0, g1, gz4, gz4, gz8, 0);
}
// data_size=1 (2B); pad: enable, every 16 DWORDs (64B) insert 4 DWORDs (16B) -> 80B row stride
#define TDM_FLAGS_A ((1u << 16) | (1u << 20) | (3u << 22) | (3u << 25))

// ---------------- fp32 -> bf16 conversion ----------------
__global__ void cvt_f32_bf16_kernel(const float* __restrict__ in, __bf16* __restrict__ out, int n) {
  int i = blockIdx.x * 256 + threadIdx.x;
  if (i < n) out[i] = f2bf(in[i]);
}

// ---------------- LayerNorm (fp32 in, bf16 out), one row per block ----------------
__global__ void ln_bf16_kernel(const float* __restrict__ x, const float* __restrict__ g,
                               const float* __restrict__ be, __bf16* __restrict__ out) {
  __shared__ float s_sum[256];
  __shared__ float s_sq[256];
  int row = blockIdx.x;
  int t = threadIdx.x;
  const float* xr = x + (size_t)row * EMBED;
  float v0 = xr[t], v1 = xr[t + 256], v2 = xr[t + 512];
  s_sum[t] = v0 + v1 + v2;
  s_sq[t]  = v0 * v0 + v1 * v1 + v2 * v2;
  __syncthreads();
#pragma unroll
  for (int off = 128; off > 0; off >>= 1) {
    if (t < off) { s_sum[t] += s_sum[t + off]; s_sq[t] += s_sq[t + off]; }
    __syncthreads();
  }
  float mu  = s_sum[0] * (1.0f / EMBED);
  float var = s_sq[0] * (1.0f / EMBED) - mu * mu;
  float rs  = rsqrtf(var + 1e-5f);
  __bf16* orow = out + (size_t)row * EMBED;
  orow[t]       = f2bf((v0 - mu) * rs * g[t]       + be[t]);
  orow[t + 256] = f2bf((v1 - mu) * rs * g[t + 256] + be[t + 256]);
  orow[t + 512] = f2bf((v2 - mu) * rs * g[t + 512] + be[t + 512]);
}

// ---------------- generic bf16 WMMA GEMM with fused epilogues ----------------
// C[M,N] = A[M,K] (bf16, row-major) * W[K,N] (bf16, row-major)
// Block tile 128(M) x 64(N), K-step 32; A tile staged by the Tensor Data Mover.
#define EPI_QKV       0
#define EPI_BIAS_RES  1
#define EPI_BIAS_GELU 2

#define LDA_S 40   // 32 + 8 pad elems: 80 B row stride (matches TDM pad config)

__global__ void gemm_bf16_kernel(const __bf16* __restrict__ A, const __bf16* __restrict__ W,
                                 const float* __restrict__ bias, const float* __restrict__ residual,
                                 float* __restrict__ outF, __bf16* __restrict__ outB,
                                 __bf16* __restrict__ qb, __bf16* __restrict__ kb, __bf16* __restrict__ vb,
                                 int M, int N, int K, int epi) {
  __shared__ __bf16 As[128 * LDA_S];
  __shared__ __bf16 Bs[64 * LDA_S];   // transposed: Bs[n][k]
  const int tid  = threadIdx.x;
  const int wave = tid >> 5;
  const int lane = tid & 31;
  const int wm = wave >> 1;          // 0..3
  const int wn = wave & 1;           // 0..1
  const int rowBase = blockIdx.y * 128;
  const int colBase = blockIdx.x * 64;
  const unsigned asOff = (unsigned)(uintptr_t)(&As[0]);

  // B-tile staging plan (hoisted out of K loop): 1024 bf16-pairs, 4 per thread.
  // pair p: kr = p>>5 (K row 0..31), nc = (p&31)*2 (even column)
  __bf16* bDst[4];
  unsigned bOff[4];   // global element offset within W minus kt*N
#pragma unroll
  for (int i = 0; i < 4; ++i) {
    int p = tid + i * 256;
    int kr = p >> 5, nc = (p & 31) * 2;
    bDst[i] = &Bs[nc * LDA_S + kr];
    bOff[i] = (unsigned)(kr * N + colBase + nc);
  }

  v8f acc[2][2];
#pragma unroll
  for (int i = 0; i < 2; ++i)
#pragma unroll
    for (int j = 0; j < 2; ++j) acc[i][j] = (v8f){0,0,0,0,0,0,0,0};

  for (int kt = 0; kt < K; kt += 32) {
    __syncthreads();
    // A tile 128x32 via TDM (wave 0 issues one DMA; pad makes 80B LDS rows)
    if (tid < 32) {
      tdm_load_2d_bf16(asOff, A + (size_t)rowBase * K + kt,
                       (unsigned)K, (unsigned)M, (unsigned)K,
                       32u, 128u, TDM_FLAGS_A);
    }
    // W tile 32 x 64, transposed into Bs[n][k]; 32-bit loads, offsets hoisted
    {
      unsigned ktN = (unsigned)kt * (unsigned)N;
#pragma unroll
      for (int i = 0; i < 4; ++i) {
        unsigned u = *(const unsigned*)(W + (size_t)(ktN + bOff[i]));
        bDst[i][0]     = u16bf((unsigned short)u);
        bDst[i][LDA_S] = u16bf((unsigned short)(u >> 16));
      }
    }
    if (tid < 32) __builtin_amdgcn_s_wait_tensorcnt(0);
    __syncthreads();

    v16bf a0 = load_a_frag(As + (wm * 32)      * LDA_S, LDA_S);
    v16bf a1 = load_a_frag(As + (wm * 32 + 16) * LDA_S, LDA_S);
    v16bf b0 = load_b_frag(Bs + (wn * 32)      * LDA_S, LDA_S);
    v16bf b1 = load_b_frag(Bs + (wn * 32 + 16) * LDA_S, LDA_S);
    acc[0][0] = wmma_bf16(a0, b0, acc[0][0]);
    acc[0][1] = wmma_bf16(a0, b1, acc[0][1]);
    acc[1][0] = wmma_bf16(a1, b0, acc[1][0]);
    acc[1][1] = wmma_bf16(a1, b1, acc[1][1]);
  }

  // ---- epilogue ----
  const int lrow = (lane >> 4) * 8;   // C layout: vgpr r -> M = r + lrow
  const int lcol = lane & 15;
#pragma unroll
  for (int tn = 0; tn < 2; ++tn) {
    int gcol = colBase + wn * 32 + tn * 16 + lcol;
    // hoisted per-column decode for QKV scatter
    int which = gcol / EMBED;
    int cc = gcol - which * EMBED;
    int hh = cc >> 6, dd = cc & 63;
#pragma unroll
    for (int tm = 0; tm < 2; ++tm)
#pragma unroll
      for (int r = 0; r < 8; ++r) {
        int grow = rowBase + wm * 32 + tm * 16 + lrow + r;
        float v = acc[tm][tn][r];
        if (epi == EPI_QKV) {
          int bb = grow >> 11, nn = grow & 2047;
          size_t dst = (((size_t)bb * HEADS + hh) * SEQ + nn) * HDIM + dd;
          float val = (which == 0) ? v * 0.125f : v;
          __bf16* dp = (which == 0) ? qb : (which == 1) ? kb : vb;
          dp[dst] = f2bf(val);
        } else if (epi == EPI_BIAS_RES) {
          size_t idx = (size_t)grow * N + gcol;
          outF[idx] = residual[idx] + v + bias[gcol];
        } else { // EPI_BIAS_GELU -> bf16
          float tv = v + bias[gcol];
          float gel = 0.5f * tv * (1.0f + erff(tv * 0.70710678118654752f));
          outB[(size_t)grow * N + gcol] = f2bf(gel);
        }
      }
  }
}

// ---------------- flash attention: softmax(q k^T) v, per (b,h) ----------------
// grid.x = SEQ/128 query tiles, grid.y = B*H. 256 threads = 8 waves x 16 query rows.
#define LQ 80   // 64 + 16 pad elems: 160 B rows, 16B aligned for b128 LDS ops
#define LV 40   // 32 + 8
__global__ void attn_kernel(const __bf16* __restrict__ qg, const __bf16* __restrict__ kg,
                            const __bf16* __restrict__ vg, __bf16* __restrict__ og) {
  __shared__ __bf16 Qs[128 * LQ];          // [qrow][d]
  __shared__ __bf16 Ks[32 * LQ];           // [krow][d]  (== B^T layout for scores)
  __shared__ __bf16 Vt[64 * LV];           // [d][krow]  (== B^T layout for PV)
  __shared__ __bf16 Ps[8 * 16 * LV];       // per-wave P tile 16x32 (A layout)

  const int tid  = threadIdx.x;
  const int wave = tid >> 5;
  const int lane = tid & 31;
  const int bh = blockIdx.y;
  const int b  = bh / HEADS, h = bh % HEADS;
  const int qbase = blockIdx.x * 128;

  const __bf16* qp = qg + (size_t)bh * SEQ * HDIM;
  const __bf16* kp = kg + (size_t)bh * SEQ * HDIM;
  const __bf16* vp = vg + (size_t)bh * SEQ * HDIM;

  // stage Q tile 128x64 as 16-byte vector copies (rows 128B in global, 160B in LDS)
#pragma unroll
  for (int i = 0; i < 4; ++i) {
    int c = tid + i * 256;               // 1024 chunks of 8 bf16
    int rr = c >> 3, d8 = (c & 7) * 8;
    *(uint4*)(&Qs[rr * LQ + d8]) =
        *(const uint4*)(&qp[(size_t)(qbase + rr) * HDIM + d8]);
  }

  // per-lane async-copy addressing for the K tile: 16 B per lane covers 32x64 bf16
  const int krr = tid >> 3;            // 0..31 (key row)
  const int kd8 = (tid & 7) * 8;       // d offset in elems (16B chunks)
  const unsigned ksLds = (unsigned)(uintptr_t)(&Ks[0]) + (unsigned)(krr * LQ + kd8) * 2u;
  const unsigned long long kGlobBase =
      (unsigned long long)(uintptr_t)(kp + (size_t)krr * HDIM + kd8);

  // V-tile staging plan (hoisted): 1024 pairs, 4 per thread; pair p: rr=p>>5, dd=(p&31)*2
  __bf16* vDst[4];
  unsigned vOff[4];
#pragma unroll
  for (int i = 0; i < 4; ++i) {
    int p = tid + i * 256;
    int rr = p >> 5, ddp = (p & 31) * 2;
    vDst[i] = &Vt[ddp * LV + rr];
    vOff[i] = (unsigned)(rr * HDIM + ddp);
  }

  v8f oacc[4];
#pragma unroll
  for (int t = 0; t < 4; ++t) oacc[t] = (v8f){0,0,0,0,0,0,0,0};
  float mrow[8], lrow[8];
#pragma unroll
  for (int r = 0; r < 8; ++r) { mrow[r] = -1e30f; lrow[r] = 0.0f; }

  __bf16* Pw = Ps + wave * 16 * LV;

  for (int kt = 0; kt < SEQ; kt += 32) {
    __syncthreads();
    // K tile: async global->LDS copy (ASYNCcnt path), 16 B per lane
    {
      unsigned long long ga = kGlobBase + (unsigned long long)kt * (HDIM * 2);
      asm volatile("global_load_async_to_lds_b128 %0, %1, off"
                   :: "v"(ksLds), "v"(ga) : "memory");
    }
    // V tile: transposed store (async copy cannot transpose); 32-bit loads
    {
      unsigned ktd = (unsigned)kt * HDIM;
#pragma unroll
      for (int i = 0; i < 4; ++i) {
        unsigned u = *(const unsigned*)(vp + (size_t)(ktd + vOff[i]));
        vDst[i][0]  = u16bf((unsigned short)u);
        vDst[i][LV] = u16bf((unsigned short)(u >> 16));
      }
    }
    asm volatile("s_wait_asynccnt 0x0" ::: "memory");
    __syncthreads();

    // scores S (16 x 32) = Q(16x64) @ K^T : two K-steps of 32
    v8f s0 = (v8f){0,0,0,0,0,0,0,0};
    v8f s1 = (v8f){0,0,0,0,0,0,0,0};
    v16bf qa0 = load_a_frag(Qs + (wave * 16) * LQ,      LQ);  // d 0..31
    v16bf qa1 = load_a_frag(Qs + (wave * 16) * LQ + 32, LQ);  // d 32..63
    v16bf kb00 = load_b_frag(Ks,                LQ);          // keys 0-15,  d 0..31
    v16bf kb01 = load_b_frag(Ks + 32,           LQ);          // keys 0-15,  d 32..63
    v16bf kb10 = load_b_frag(Ks + 16 * LQ,      LQ);          // keys 16-31, d 0..31
    v16bf kb11 = load_b_frag(Ks + 16 * LQ + 32, LQ);          // keys 16-31, d 32..63
    s0 = wmma_bf16(qa0, kb00, s0); s0 = wmma_bf16(qa1, kb01, s0);
    s1 = wmma_bf16(qa0, kb10, s1); s1 = wmma_bf16(qa1, kb11, s1);

    // online softmax (row stats per 16-lane half; row M = r + 8*(lane>=16))
#pragma unroll
    for (int r = 0; r < 8; ++r) {
      float mx = fmaxf(s0[r], s1[r]);
#pragma unroll
      for (int off = 8; off >= 1; off >>= 1) mx = fmaxf(mx, __shfl_xor(mx, off, 32));
      float mnew = fmaxf(mrow[r], mx);
      float scale = __expf(mrow[r] - mnew);
      float p0 = __expf(s0[r] - mnew);
      float p1 = __expf(s1[r] - mnew);
      float ps = p0 + p1;
#pragma unroll
      for (int off = 8; off >= 1; off >>= 1) ps += __shfl_xor(ps, off, 32);
      lrow[r] = lrow[r] * scale + ps;
      mrow[r] = mnew;
#pragma unroll
      for (int t = 0; t < 4; ++t) oacc[t][r] *= scale;
      int prow = r + ((lane >> 4) << 3);
      Pw[prow * LV + (lane & 15)]      = f2bf(p0);
      Pw[prow * LV + 16 + (lane & 15)] = f2bf(p1);
    }
    // per-wave LDS ops are in-order: safe to reload P as A fragment
    v16bf pa = load_a_frag(Pw, LV);
#pragma unroll
    for (int t = 0; t < 4; ++t) {
      v16bf vf = load_b_frag(Vt + (t * 16) * LV, LV);   // N = d tile, K = 32 keys
      oacc[t] = wmma_bf16(pa, vf, oacc[t]);
    }
  }

  // write O (bf16) to [B, N, C] with C = h*64 + d
#pragma unroll
  for (int t = 0; t < 4; ++t)
#pragma unroll
    for (int r = 0; r < 8; ++r) {
      int prow = r + ((lane >> 4) << 3);
      int grow = qbase + wave * 16 + prow;
      int dd = t * 16 + (lane & 15);
      float val = oacc[t][r] / lrow[r];
      og[((size_t)b * SEQ + grow) * EMBED + h * HDIM + dd] = f2bf(val);
    }
}

// ---------------- host launcher ----------------
extern "C" void kernel_launch(void* const* d_in, const int* in_sizes, int n_in,
                              void* d_out, int out_size, void* d_ws, size_t ws_size,
                              hipStream_t stream) {
  const float* x      = (const float*)d_in[0];
  const float* ln1_g  = (const float*)d_in[1];
  const float* ln1_b  = (const float*)d_in[2];
  const float* w_qkv  = (const float*)d_in[3];
  const float* w_proj = (const float*)d_in[4];
  const float* b_proj = (const float*)d_in[5];
  const float* ln2_g  = (const float*)d_in[6];
  const float* ln2_b  = (const float*)d_in[7];
  const float* w1     = (const float*)d_in[8];
  const float* b1     = (const float*)d_in[9];
  const float* w2     = (const float*)d_in[10];
  const float* b2     = (const float*)d_in[11];
  float* out = (float*)d_out;

  // workspace carve-up (bf16 buffers, 256B aligned)
  char* ws = (char*)d_ws;
  size_t off = 0;
  auto alloc = [&](size_t bytes) -> void* {
    void* p = ws + off;
    off += (bytes + 255) & ~(size_t)255;
    return p;
  };
  __bf16* wqkv_bf  = (__bf16*)alloc((size_t)EMBED * 3 * EMBED * 2);
  __bf16* wproj_bf = (__bf16*)alloc((size_t)EMBED * EMBED * 2);
  __bf16* w1_bf    = (__bf16*)alloc((size_t)EMBED * DFF * 2);
  __bf16* w2_bf    = (__bf16*)alloc((size_t)DFF * EMBED * 2);
  __bf16* h_bf     = (__bf16*)alloc((size_t)ROWS * EMBED * 2);   // LN outputs (reused)
  __bf16* q_bf     = (__bf16*)alloc((size_t)ROWS * EMBED * 2);   // [B,H,N,D]
  __bf16* k_bf     = (__bf16*)alloc((size_t)ROWS * EMBED * 2);
  __bf16* v_bf     = (__bf16*)alloc((size_t)ROWS * EMBED * 2);
  __bf16* o_bf     = (__bf16*)alloc((size_t)ROWS * EMBED * 2);   // [B,N,C]
  __bf16* h1_bf    = (__bf16*)alloc((size_t)ROWS * DFF * 2);     // MLP hidden

  // 1) weights -> bf16
  cvt_f32_bf16_kernel<<<(EMBED * 3 * EMBED) / 256, 256, 0, stream>>>(w_qkv, wqkv_bf, EMBED * 3 * EMBED);
  cvt_f32_bf16_kernel<<<(EMBED * EMBED) / 256,     256, 0, stream>>>(w_proj, wproj_bf, EMBED * EMBED);
  cvt_f32_bf16_kernel<<<(EMBED * DFF) / 256,       256, 0, stream>>>(w1, w1_bf, EMBED * DFF);
  cvt_f32_bf16_kernel<<<(DFF * EMBED) / 256,       256, 0, stream>>>(w2, w2_bf, DFF * EMBED);

  // 2) LN1
  ln_bf16_kernel<<<ROWS, 256, 0, stream>>>(x, ln1_g, ln1_b, h_bf);

  // 3) QKV GEMM (8192 x 2304 x 768) -> q/k/v [B,H,N,D], q pre-scaled by 1/8
  {
    dim3 grid((3 * EMBED) / 64, ROWS / 128);
    gemm_bf16_kernel<<<grid, 256, 0, stream>>>(h_bf, wqkv_bf, nullptr, nullptr,
                                               nullptr, nullptr, q_bf, k_bf, v_bf,
                                               ROWS, 3 * EMBED, EMBED, EPI_QKV);
  }

  // 4) attention -> o_bf [B,N,C]
  {
    dim3 grid(SEQ / 128, BATCH * HEADS);
    attn_kernel<<<grid, 256, 0, stream>>>(q_bf, k_bf, v_bf, o_bf);
  }

  // 5) proj GEMM + bias + residual(x) -> out (fp32)
  {
    dim3 grid(EMBED / 64, ROWS / 128);
    gemm_bf16_kernel<<<grid, 256, 0, stream>>>(o_bf, wproj_bf, b_proj, x,
                                               out, nullptr, nullptr, nullptr, nullptr,
                                               ROWS, EMBED, EMBED, EPI_BIAS_RES);
  }

  // 6) LN2 on out -> h_bf
  ln_bf16_kernel<<<ROWS, 256, 0, stream>>>(out, ln2_g, ln2_b, h_bf);

  // 7) FF1 + bias + GELU -> h1_bf
  {
    dim3 grid(DFF / 64, ROWS / 128);
    gemm_bf16_kernel<<<grid, 256, 0, stream>>>(h_bf, w1_bf, b1, nullptr,
                                               nullptr, h1_bf, nullptr, nullptr, nullptr,
                                               ROWS, DFF, EMBED, EPI_BIAS_GELU);
  }

  // 8) FF2 + bias + residual(out) -> out (in-place safe: 1:1 element mapping)
  {
    dim3 grid(EMBED / 64, ROWS / 128);
    gemm_bf16_kernel<<<grid, 256, 0, stream>>>(h1_bf, w2_bf, b2, out,
                                               out, nullptr, nullptr, nullptr, nullptr,
                                               ROWS, EMBED, DFF, EPI_BIAS_RES);
  }
}
